// AttentionModel_63737314672806
// MI455X (gfx1250) — compile-verified, matrix operand order checked by reading, not run on hardware
//
#include <hip/hip_runtime.h>
#include <hip/hip_bf16.h>
#include <stdint.h>

// ---------------------------------------------------------------------------
// Types for WMMA (CDNA5 / gfx1250, wave32): bf16 A/B fragments, f32 accum.
// ---------------------------------------------------------------------------
typedef __attribute__((ext_vector_type(16))) __bf16       v16bf;
typedef __attribute__((ext_vector_type(8)))  float        v8f;
typedef __attribute__((ext_vector_type(4)))  unsigned int v4u;   // trivially copyable 16B chunk

union U4   { v4u q; unsigned short s[8]; };
union Frag { v4u u[2]; v16bf v; };

// CDNA5 async global->LDS (ASYNCcnt) path, guarded so absence can't break compile.
#if defined(__AMDGCN__) && __has_builtin(__builtin_amdgcn_global_load_async_to_lds_b128)
#define HAVE_ASYNC 1
#else
#define HAVE_ASYNC 0
#endif

#if HAVE_ASYNC
// Builtin signature (from hipcc diagnostics): pointee is a GCC-style int4 vector.
typedef int v4i_ __attribute__((vector_size(16)));
#define GAS __attribute__((address_space(1)))
#define LAS __attribute__((address_space(3)))
// generic->AS1: flat and global addresses coincide on amdgcn.
#define TO_GLB(p) ((GAS v4i_*)(uintptr_t)(const void*)(p))
// generic->AS3: low 32 bits of a flat shared address are the LDS byte offset.
#define TO_LDS(p) ((LAS v4i_*)(unsigned int)(uintptr_t)(const void*)(p))
__device__ __forceinline__ void wait_async0() {
#if __has_builtin(__builtin_amdgcn_s_wait_asynccnt)
  __builtin_amdgcn_s_wait_asynccnt(0);
#else
  asm volatile("s_wait_asynccnt 0x0" ::: "memory");
#endif
}
#endif

__device__ __forceinline__ unsigned short f2bf(float f) {
  unsigned int u = __float_as_uint(f);
  u += 0x7FFFu + ((u >> 16) & 1u);          // round-to-nearest-even
  return (unsigned short)(u >> 16);
}
__device__ __forceinline__ float bf2f(unsigned short h) {
  return __uint_as_float(((unsigned int)h) << 16);
}
__device__ __forceinline__ float sigmoidf_(float x) { return 1.0f / (1.0f + __expf(-x)); }

__device__ __forceinline__ v8f wmma_bf16(const Frag& a, const Frag& b, v8f c) {
  // (neg_a, A, neg_b, B, c_mod, C, reuse_a, reuse_b)
  return __builtin_amdgcn_wmma_f32_16x16x32_bf16(false, a.v, false, b.v, (short)0, c, false, false);
}

#define SA 40   // padded LDS stride (bf16 elems): 80B rows, 16B aligned, conflict-breaking

// ---------------------------------------------------------------------------
// f32 -> bf16 conversion
// ---------------------------------------------------------------------------
__global__ void __launch_bounds__(256) k_cvt(const float* __restrict__ src,
                                             unsigned short* __restrict__ dst, int n) {
  int i = blockIdx.x * 256 + threadIdx.x;
  if (i < n) dst[i] = f2bf(src[i]);
}

// ---------------------------------------------------------------------------
// Tiled WMMA GEMM: C[M,N] = act(A[M,K] @ W[K,N] + bias).
// Block tile 128x128, BK=32, 256 threads = 8 waves (2Mx4N), wave tile 64x32.
// Double-buffered LDS; A tile staged via async global->LDS when available.
// act: 0 = bias, 1 = sigmoid(bias+acc), 3 = Cprev * sigmoid(bias+acc)
// ---------------------------------------------------------------------------
__global__ void __launch_bounds__(256) k_gemm(
    const unsigned short* __restrict__ A, int lda,
    const unsigned short* __restrict__ Bm, int ldb,
    const float* __restrict__ bias,
    unsigned short* __restrict__ C, int ldc,
    int K, int act)
{
  __shared__ unsigned short As[2][128 * SA];   // A tile [row][k]
  __shared__ unsigned short Bs[2][128 * SA];   // B tile transposed [n][k]
  const int tid  = threadIdx.x;
  const int lane = tid & 31, wv = tid >> 5;
  const int wm = wv >> 2, wn = wv & 3;
  const int lr = lane & 15, lh = lane >> 4;
  const int mBase = blockIdx.y * 128;
  const int nBase = blockIdx.x * 128;

  // Per-thread staging coordinates (two 16B chunks each for A and B tiles).
  const int ar0 = tid >> 2,          ac0 = (tid & 3) * 8;
  const int ar1 = (tid >> 2) + 64;   // (tid+256)>>2 ; same column chunk
  const int bk0 = tid >> 4,          bn0 = (tid & 15) * 8;
  const int bk1 = (tid >> 4) + 16;   // (tid+256)>>4 ; same column chunk

  v8f acc[4][2];
  #pragma unroll
  for (int i = 0; i < 4; ++i)
    #pragma unroll
    for (int j = 0; j < 2; ++j) acc[i][j] = (v8f){0, 0, 0, 0, 0, 0, 0, 0};

  U4 bu0, bu1;

  auto stageA = [&](int buf, int k0) {
#if HAVE_ASYNC
    __builtin_amdgcn_global_load_async_to_lds_b128(
        TO_GLB(A + (size_t)(mBase + ar0) * lda + k0 + ac0),
        TO_LDS(&As[buf][ar0 * SA + ac0]), 0, 0);
    __builtin_amdgcn_global_load_async_to_lds_b128(
        TO_GLB(A + (size_t)(mBase + ar1) * lda + k0 + ac0),
        TO_LDS(&As[buf][ar1 * SA + ac0]), 0, 0);
#else
    v4u t0 = *(const v4u*)(A + (size_t)(mBase + ar0) * lda + k0 + ac0);
    v4u t1 = *(const v4u*)(A + (size_t)(mBase + ar1) * lda + k0 + ac0);
    *(v4u*)(&As[buf][ar0 * SA + ac0]) = t0;
    *(v4u*)(&As[buf][ar1 * SA + ac0]) = t1;
#endif
  };
  auto loadB = [&](int k0) {
    bu0.q = *(const v4u*)(Bm + (size_t)(k0 + bk0) * ldb + nBase + bn0);
    bu1.q = *(const v4u*)(Bm + (size_t)(k0 + bk1) * ldb + nBase + bn0);
  };
  auto storeB = [&](int buf) {
    #pragma unroll
    for (int j = 0; j < 8; ++j) Bs[buf][(bn0 + j) * SA + bk0] = bu0.s[j];
    #pragma unroll
    for (int j = 0; j < 8; ++j) Bs[buf][(bn0 + j) * SA + bk1] = bu1.s[j];
  };
  auto compute = [&](int buf) {
    Frag a[4], b[2];
    #pragma unroll
    for (int mt = 0; mt < 4; ++mt) {
      int row = wm * 64 + mt * 16 + lr;
      a[mt].u[0] = *(const v4u*)(&As[buf][row * SA + lh * 16]);
      a[mt].u[1] = *(const v4u*)(&As[buf][row * SA + lh * 16 + 8]);
    }
    #pragma unroll
    for (int nt = 0; nt < 2; ++nt) {
      int col = wn * 32 + nt * 16 + lr;
      b[nt].u[0] = *(const v4u*)(&Bs[buf][col * SA + lh * 16]);
      b[nt].u[1] = *(const v4u*)(&Bs[buf][col * SA + lh * 16 + 8]);
    }
    #pragma unroll
    for (int mt = 0; mt < 4; ++mt)
      #pragma unroll
      for (int nt = 0; nt < 2; ++nt)
        acc[mt][nt] = wmma_bf16(a[mt], b[nt], acc[mt][nt]);
  };

  // Prologue: tile 0 into buffer 0.
  stageA(0, 0);
  loadB(0);
  storeB(0);
#if HAVE_ASYNC
  wait_async0();
#endif
  __syncthreads();

  const int nk = K >> 5;
  for (int ki = 0; ki < nk; ++ki) {
    const int buf = ki & 1;
    if (ki + 1 < nk) {                 // issue next tile before computing
      stageA(buf ^ 1, (ki + 1) << 5);  // async writes land in the other buffer
      loadB((ki + 1) << 5);            // B chunks parked in registers
    }
    compute(buf);
    if (ki + 1 < nk) {
      storeB(buf ^ 1);
#if HAVE_ASYNC
      wait_async0();
#endif
    }
    __syncthreads();
  }

  // Epilogue: C/D layout — VGPR r: M=r (lanes 0-15), M=r+8 (lanes 16-31); N=lane%16
  #pragma unroll
  for (int mt = 0; mt < 4; ++mt) {
    int row0 = mBase + wm * 64 + mt * 16 + lh * 8;
    #pragma unroll
    for (int nt = 0; nt < 2; ++nt) {
      int col = nBase + wn * 32 + nt * 16 + lr;
      float bb = bias[col];
      #pragma unroll
      for (int r = 0; r < 8; ++r) {
        size_t idx = (size_t)(row0 + r) * ldc + col;
        float val = acc[mt][nt][r] + bb;
        if (act == 1)      val = sigmoidf_(val);
        else if (act == 3) val = bf2f(C[idx]) * sigmoidf_(val);
        C[idx] = f2bf(val);
      }
    }
  }
}

// ---------------------------------------------------------------------------
// Attention logits + softmax per (b,h).
// S = Q Kt / sqrt(dk) + rel[(b-q+127), k]; softmax over k; P bf16.
// K_bh is staged once into LDS (async when available); fragments need only
// contiguous runs along DK, so no transpose is required.
// ---------------------------------------------------------------------------
__global__ void __launch_bounds__(256) k_attn_logits(
    const unsigned short* __restrict__ q, const unsigned short* __restrict__ k,
    const float* __restrict__ rel, unsigned short* __restrict__ P)
{
  __shared__ unsigned short Ks[128 * 136];   // Ks[key_row][dk], padded stride
  int bh = blockIdx.x; int b = bh >> 3, h = bh & 7;
  int tid = threadIdx.x, lane = tid & 31, wv = tid >> 5;
  int lr = lane & 15, lh = lane >> 4;
  const unsigned short* qb = q + (size_t)b * 131072 + h * 128;
  const unsigned short* kb = k + (size_t)b * 131072 + h * 128;

#if HAVE_ASYNC
  #pragma unroll
  for (int it = 0; it < 8; ++it) {
    int idx = it * 256 + tid;              // 2048 x 16B chunks
    int kr = idx >> 4, nc = (idx & 15) * 8;
    __builtin_amdgcn_global_load_async_to_lds_b128(
        TO_GLB(kb + (size_t)kr * 1024 + nc), TO_LDS(&Ks[kr * 136 + nc]), 0, 0);
  }
  wait_async0();
#else
  v4u stg[8];
  #pragma unroll
  for (int it = 0; it < 8; ++it) {
    int idx = it * 256 + tid;
    int kr = idx >> 4, nc = (idx & 15) * 8;
    stg[it] = *(const v4u*)(kb + (size_t)kr * 1024 + nc);
  }
  #pragma unroll
  for (int it = 0; it < 8; ++it) {
    int idx = it * 256 + tid;
    int kr = idx >> 4, nc = (idx & 15) * 8;
    *(v4u*)(&Ks[kr * 136 + nc]) = stg[it];
  }
#endif
  __syncthreads();

  v8f acc[8];
  #pragma unroll
  for (int nt = 0; nt < 8; ++nt) acc[nt] = (v8f){0, 0, 0, 0, 0, 0, 0, 0};

  int rowq = wv * 16 + lr;
  for (int ks = 0; ks < 128; ks += 32) {
    Frag a;
    const unsigned short* ap = qb + (size_t)rowq * 1024 + ks + lh * 16;
    a.u[0] = *(const v4u*)ap;  a.u[1] = *(const v4u*)(ap + 8);
    #pragma unroll
    for (int nt = 0; nt < 8; ++nt) {
      Frag bf; int colk = nt * 16 + lr;
      const unsigned short* bp = &Ks[colk * 136 + ks + lh * 16];
      bf.u[0] = *(const v4u*)bp;  bf.u[1] = *(const v4u*)(bp + 8);
      acc[nt] = wmma_bf16(a, bf, acc[nt]);
    }
  }

  const float sc = 0.08838834764831845f;  // 1/sqrt(128)
  #pragma unroll
  for (int nt = 0; nt < 8; ++nt) {
    int colk = nt * 16 + lr;
    #pragma unroll
    for (int r = 0; r < 8; ++r) {
      int rq = wv * 16 + r + lh * 8;
      acc[nt][r] = acc[nt][r] * sc + rel[(b - rq + 127) * 128 + colk];
    }
  }
  // softmax per row: 128 values live across 16 lanes (half-wave) x 8 tiles
  #pragma unroll
  for (int r = 0; r < 8; ++r) {
    float m = -1e30f;
    #pragma unroll
    for (int nt = 0; nt < 8; ++nt) m = fmaxf(m, acc[nt][r]);
    for (int msk = 1; msk < 16; msk <<= 1) m = fmaxf(m, __shfl_xor(m, msk, 32));
    float s = 0.f;
    #pragma unroll
    for (int nt = 0; nt < 8; ++nt) { float e = __expf(acc[nt][r] - m); acc[nt][r] = e; s += e; }
    for (int msk = 1; msk < 16; msk <<= 1) s += __shfl_xor(s, msk, 32);
    float inv = 1.0f / s;
    int rq = wv * 16 + r + lh * 8;
    unsigned short* pr = P + ((size_t)bh * 128 + rq) * 128;
    #pragma unroll
    for (int nt = 0; nt < 8; ++nt) pr[nt * 16 + lr] = f2bf(acc[nt][r] * inv);
  }
}

// ---------------------------------------------------------------------------
// ao[b,q,h,:] = P[b,h,q,:] @ V[b,:,h,:]  — V transposed through LDS.
// ---------------------------------------------------------------------------
__global__ void __launch_bounds__(256) k_attn_av(
    const unsigned short* __restrict__ P, const unsigned short* __restrict__ v,
    unsigned short* __restrict__ ao)
{
  __shared__ unsigned short Vt[128 * 136];   // Vt[n][k], padded stride 136
  int bh = blockIdx.x, b = bh >> 3, h = bh & 7;
  int tid = threadIdx.x, lane = tid & 31, wv = tid >> 5;
  int lr = lane & 15, lh = lane >> 4;
  const unsigned short* vb = v + (size_t)b * 131072 + h * 128;

  // Two-phase staging: batch the global loads, then scatter-transpose into LDS.
  U4 stg[8];
  #pragma unroll
  for (int it = 0; it < 8; ++it) {
    int idx = it * 256 + tid;              // 2048 x 16B chunks
    int kr = idx >> 4, nc = (idx & 15) * 8;
    stg[it].q = *(const v4u*)(vb + (size_t)kr * 1024 + nc);
  }
  #pragma unroll
  for (int it = 0; it < 8; ++it) {
    int idx = it * 256 + tid;
    int kr = idx >> 4, nc = (idx & 15) * 8;
    #pragma unroll
    for (int j = 0; j < 8; ++j) Vt[(nc + j) * 136 + kr] = stg[it].s[j];
  }
  __syncthreads();

  v8f acc[8];
  #pragma unroll
  for (int nt = 0; nt < 8; ++nt) acc[nt] = (v8f){0, 0, 0, 0, 0, 0, 0, 0};

  const unsigned short* pb = P + (size_t)bh * 16384;
  int rowq = wv * 16 + lr;
  for (int ks = 0; ks < 128; ks += 32) {
    Frag a;
    const unsigned short* ap = pb + (size_t)rowq * 128 + ks + lh * 16;
    a.u[0] = *(const v4u*)ap;  a.u[1] = *(const v4u*)(ap + 8);
    #pragma unroll
    for (int nt = 0; nt < 8; ++nt) {
      Frag bf;
      const unsigned short* bp = &Vt[(nt * 16 + lr) * 136 + ks + lh * 16];
      bf.u[0] = *(const v4u*)bp;  bf.u[1] = *(const v4u*)(bp + 8);
      acc[nt] = wmma_bf16(a, bf, acc[nt]);
    }
  }
  #pragma unroll
  for (int nt = 0; nt < 8; ++nt) {
    int col = nt * 16 + lr;
    #pragma unroll
    for (int r = 0; r < 8; ++r) {
      int rq = wv * 16 + r + lh * 8;
      ao[(size_t)b * 131072 + (size_t)rq * 1024 + h * 128 + col] = f2bf(acc[nt][r]);
    }
  }
}

// ---------------------------------------------------------------------------
// h = LN(x + gate * a) over D=1024, one row per block.
// ---------------------------------------------------------------------------
__global__ void __launch_bounds__(256) k_gate_res_ln(
    const unsigned short* __restrict__ x, const unsigned short* __restrict__ a,
    const unsigned short* __restrict__ g, const float* __restrict__ s,
    const float* __restrict__ bta, unsigned short* __restrict__ out)
{
  __shared__ float red[256], red2[256];
  size_t base = (size_t)blockIdx.x * 1024;
  int tid = threadIdx.x;
  float t[4]; float sum = 0.f, sq = 0.f;
  #pragma unroll
  for (int j = 0; j < 4; ++j) {
    int d = tid + j * 256;
    float val = bf2f(x[base + d]) + bf2f(g[base + d]) * bf2f(a[base + d]);
    t[j] = val; sum += val; sq += val * val;
  }
  red[tid] = sum; red2[tid] = sq; __syncthreads();
  for (int o = 128; o > 0; o >>= 1) {
    if (tid < o) { red[tid] += red[tid + o]; red2[tid] += red2[tid + o]; }
    __syncthreads();
  }
  float mean = red[0] * (1.f / 1024.f);
  float var  = red2[0] * (1.f / 1024.f) - mean * mean;
  float rinv = rsqrtf(var + 1e-6f);
  #pragma unroll
  for (int j = 0; j < 4; ++j) {
    int d = tid + j * 256;
    out[base + d] = f2bf((t[j] - mean) * rinv * s[d] + bta[d]);
  }
}

// ---------------------------------------------------------------------------
// blk = LN(h + f2); x = x + surv*(blk - x)   (in-place on x)
// ---------------------------------------------------------------------------
__global__ void __launch_bounds__(256) k_res_ln_mix(
    unsigned short* __restrict__ x, const unsigned short* __restrict__ h,
    const unsigned short* __restrict__ f2, const float* __restrict__ s,
    const float* __restrict__ bta, float surv)
{
  __shared__ float red[256], red2[256];
  size_t base = (size_t)blockIdx.x * 1024;
  int tid = threadIdx.x;
  float t[4]; float sum = 0.f, sq = 0.f;
  #pragma unroll
  for (int j = 0; j < 4; ++j) {
    int d = tid + j * 256;
    float val = bf2f(h[base + d]) + bf2f(f2[base + d]);
    t[j] = val; sum += val; sq += val * val;
  }
  red[tid] = sum; red2[tid] = sq; __syncthreads();
  for (int o = 128; o > 0; o >>= 1) {
    if (tid < o) { red[tid] += red[tid + o]; red2[tid] += red2[tid + o]; }
    __syncthreads();
  }
  float mean = red[0] * (1.f / 1024.f);
  float var  = red2[0] * (1.f / 1024.f) - mean * mean;
  float rinv = rsqrtf(var + 1e-6f);
  #pragma unroll
  for (int j = 0; j < 4; ++j) {
    int d = tid + j * 256;
    float blk = (t[j] - mean) * rinv * s[d] + bta[d];
    float xo  = bf2f(x[base + d]);
    x[base + d] = f2bf(xo + surv * (blk - xo));
  }
}

// ---------------------------------------------------------------------------
// pooled[b] = [mean_l x, max_l x, other]  -> [128, 2112] f32
// ---------------------------------------------------------------------------
__global__ void __launch_bounds__(256) k_pool(
    const unsigned short* __restrict__ x, const float* __restrict__ other,
    float* __restrict__ pooled)
{
  int b = blockIdx.x, tid = threadIdx.x;
  for (int d = tid; d < 1024; d += 256) {
    const unsigned short* p = x + (size_t)b * 131072 + d;
    float m = 0.f, mx = -1e30f;
    for (int l = 0; l < 128; ++l) { float vv = bf2f(p[l * 1024]); m += vv; mx = fmaxf(mx, vv); }
    pooled[b * 2112 + d]        = m * (1.f / 128.f);
    pooled[b * 2112 + 1024 + d] = mx;
  }
  for (int j = tid; j < 64; j += 256) pooled[b * 2112 + 2048 + j] = other[b * 64 + j];
}

// ---------------------------------------------------------------------------
// Head: y = relu(pooled@Wd1+b); y = LN(y); y = relu(y@Wd2+b); out = y@Wout+bout
// ---------------------------------------------------------------------------
__global__ void __launch_bounds__(128) k_head(
    const float* __restrict__ pooled, const float* __restrict__ Wd1, const float* __restrict__ bd1,
    const float* __restrict__ s3, const float* __restrict__ b3,
    const float* __restrict__ Wd2, const float* __restrict__ bd2,
    const float* __restrict__ Wout, const float* __restrict__ bout, float* __restrict__ out)
{
  __shared__ float y1[128], rs[128], rq[128];
  int b = blockIdx.x, j = threadIdx.x;
  const float* pr = pooled + b * 2112;
  float acc = bd1[j];
  for (int i = 0; i < 2112; ++i) acc += pr[i] * Wd1[i * 128 + j];
  acc = fmaxf(acc, 0.f);
  rs[j] = acc; rq[j] = acc * acc; __syncthreads();
  for (int o = 64; o > 0; o >>= 1) {
    if (j < o) { rs[j] += rs[j + o]; rq[j] += rq[j + o]; }
    __syncthreads();
  }
  float mean = rs[0] * (1.f / 128.f);
  float var  = rq[0] * (1.f / 128.f) - mean * mean;
  float rinv = rsqrtf(var + 1e-6f);
  float ln = (acc - mean) * rinv * s3[j] + b3[j];
  y1[j] = ln; __syncthreads();
  float a2 = bd2[j];
  for (int k2 = 0; k2 < 128; ++k2) a2 += y1[k2] * Wd2[k2 * 128 + j];
  a2 = fmaxf(a2, 0.f);
  rs[j] = a2 * Wout[j]; __syncthreads();
  for (int o = 64; o > 0; o >>= 1) { if (j < o) rs[j] += rs[j + o]; __syncthreads(); }
  if (j == 0) out[b] = rs[0] + bout[0];
}

// ---------------------------------------------------------------------------
extern "C" void kernel_launch(void* const* d_in, const int* in_sizes, int n_in,
                              void* d_out, int out_size, void* d_ws, size_t ws_size,
                              hipStream_t stream) {
  (void)in_sizes; (void)n_in; (void)out_size; (void)ws_size;
  const float* cgm  = (const float*)d_in[0];
  const float* other= (const float*)d_in[1];
  const float* W_in = (const float*)d_in[2];
  const float* b_in = (const float*)d_in[3];
  const float* rel  = (const float*)d_in[4];
  const float* Wq   = (const float*)d_in[5];
  const float* bq   = (const float*)d_in[6];
  const float* Wk   = (const float*)d_in[7];
  const float* bk   = (const float*)d_in[8];
  const float* Wv   = (const float*)d_in[9];
  const float* bv   = (const float*)d_in[10];
  const float* Wo   = (const float*)d_in[11];
  const float* bo   = (const float*)d_in[12];
  const float* Wg   = (const float*)d_in[13];
  const float* bg   = (const float*)d_in[14];
  const float* ln1s = (const float*)d_in[15];
  const float* ln1b = (const float*)d_in[16];
  const float* Wf1  = (const float*)d_in[17];
  const float* bf1  = (const float*)d_in[18];
  const float* Wfg  = (const float*)d_in[19];
  const float* bfg  = (const float*)d_in[20];
  const float* Wf2  = (const float*)d_in[21];
  const float* bf2  = (const float*)d_in[22];
  const float* ln2s = (const float*)d_in[23];
  const float* ln2b = (const float*)d_in[24];
  const float* Wd1  = (const float*)d_in[25];
  const float* bd1  = (const float*)d_in[26];
  const float* ln3s = (const float*)d_in[27];
  const float* ln3b = (const float*)d_in[28];
  const float* Wd2  = (const float*)d_in[29];
  const float* bd2  = (const float*)d_in[30];
  const float* Wout = (const float*)d_in[31];
  const float* bout = (const float*)d_in[32];
  float* out = (float*)d_out;

  // Workspace carve-up (all 256B aligned)
  char* base = (char*)d_ws; size_t off = 0;
  auto take = [&](size_t bytes) -> void* {
    void* p = base + off; off = (off + bytes + 255) & ~(size_t)255; return p;
  };
  const size_t NT = 16384ULL * 1024ULL;
  unsigned short* XB  = (unsigned short*)take(NT * 2);           // x (bf16)
  unsigned short* HB  = (unsigned short*)take(NT * 2);           // h
  unsigned short* B0  = (unsigned short*)take(NT * 2);           // q / wo_out
  unsigned short* B1  = (unsigned short*)take(NT * 2);           // k / gate
  unsigned short* B2  = (unsigned short*)take(NT * 2);           // v / f2out
  unsigned short* B3  = (unsigned short*)take(NT * 2);           // ao
  unsigned short* PB  = (unsigned short*)take(16777216ULL * 2);  // probs [B,H,L,L]
  unsigned short* FB  = (unsigned short*)take(16384ULL * 4096ULL * 2); // FFN mid
  unsigned short* CGB = (unsigned short*)take(16384ULL * 32ULL * 2);
  unsigned short* WS  = (unsigned short*)take(4096ULL * 1024ULL * 2); // weight scratch
  float* PD = (float*)take(128ULL * 2112ULL * 4);

  auto cvt = [&](const float* s, unsigned short* d, int n) {
    k_cvt<<<(n + 255) / 256, 256, 0, stream>>>(s, d, n);
  };
  auto gemm = [&](const unsigned short* A, int lda, const unsigned short* W, int ldb,
                  const float* bias, unsigned short* C, int ldc, int M, int N, int K, int act) {
    dim3 g(N / 128, M / 128);
    k_gemm<<<g, 256, 0, stream>>>(A, lda, W, ldb, bias, C, ldc, K, act);
  };

  // x = cgm @ W_in + b_in
  cvt(cgm, CGB, 16384 * 32);
  cvt(W_in, WS, 32 * 1024);
  gemm(CGB, 32, WS, 1024, b_in, XB, 1024, 16384, 1024, 32, 0);

  for (int i = 0; i < 2; ++i) {
    const size_t wo1 = (size_t)i * 1024 * 1024;
    const size_t wo4 = (size_t)i * 1024 * 4096;
    cvt(Wq + wo1, WS, 1048576); gemm(XB, 1024, WS, 1024, bq + i * 1024, B0, 1024, 16384, 1024, 1024, 0);
    cvt(Wk + wo1, WS, 1048576); gemm(XB, 1024, WS, 1024, bk + i * 1024, B1, 1024, 16384, 1024, 1024, 0);
    cvt(Wv + wo1, WS, 1048576); gemm(XB, 1024, WS, 1024, bv + i * 1024, B2, 1024, 16384, 1024, 1024, 0);
    k_attn_logits<<<1024, 256, 0, stream>>>(B0, B1, rel + (size_t)i * 255 * 128, PB);
    k_attn_av<<<1024, 256, 0, stream>>>(PB, B2, B3);
    cvt(Wo + wo1, WS, 1048576); gemm(B3, 1024, WS, 1024, bo + i * 1024, B0, 1024, 16384, 1024, 1024, 0);
    cvt(Wg + wo1, WS, 1048576); gemm(XB, 1024, WS, 1024, bg + i * 1024, B1, 1024, 16384, 1024, 1024, 1);
    k_gate_res_ln<<<16384, 256, 0, stream>>>(XB, B0, B1, ln1s + i * 1024, ln1b + i * 1024, HB);
    cvt(Wf1 + wo4, WS, 4194304); gemm(HB, 1024, WS, 4096, bf1 + i * 4096, FB, 4096, 16384, 4096, 1024, 0);
    cvt(Wfg + wo4, WS, 4194304); gemm(HB, 1024, WS, 4096, bfg + i * 4096, FB, 4096, 16384, 4096, 1024, 3);
    cvt(Wf2 + wo4, WS, 4194304); gemm(FB, 4096, WS, 1024, bf2 + i * 1024, B2, 1024, 16384, 1024, 4096, 0);
    float surv = (i == 0) ? 1.0f : 0.5f;
    k_res_ln_mix<<<16384, 256, 0, stream>>>(XB, HB, B2, ln2s + i * 1024, ln2b + i * 1024, surv);
  }

  k_pool<<<128, 256, 0, stream>>>(XB, other, PD);
  k_head<<<128, 128, 0, stream>>>(PD, Wd1, bd1, ln3s, ln3b, Wd2, bd2, Wout, bout, out);
}
